// DensePool_56495999811585
// MI455X (gfx1250) — 8-bit matrix operand order repaired and checked by reading, compiled, not run on hardware
//
#include <hip/hip_runtime.h>
#include <stdint.h>

#define B_ 16
#define N_ 1024
#define F_ 128

typedef int   v8i __attribute__((ext_vector_type(8)));
typedef float v8f __attribute__((ext_vector_type(8)));
typedef float v2f __attribute__((ext_vector_type(2)));

__global__ void k_adj_to_u8(const float* __restrict__ adj, uint8_t* __restrict__ out,
                            int total4) {
    int i = blockIdx.x * blockDim.x + threadIdx.x;
    if (i >= total4) return;
    const float4 v = reinterpret_cast<const float4*>(adj)[i];
    uchar4 o;
    o.x = (v.x != 0.0f); o.y = (v.y != 0.0f);
    o.z = (v.z != 0.0f); o.w = (v.w != 0.0f);
    reinterpret_cast<uchar4*>(out)[i] = o;
}

__global__ void k_adj2_wmma_iu8(const uint8_t* __restrict__ A8,
                                float* __restrict__ adjS) {
    const int lane = threadIdx.x & 31;
    const int wid  = threadIdx.x >> 5;
    const int tile = blockIdx.x * 8 + wid;
    const int tm   = (tile >> 6) << 4;
    const int tn   = (tile & 63) << 4;
    const int b    = blockIdx.y;
    const int l    = lane & 15;
    const int half = lane >> 4;

    const uint8_t* base = A8 + (size_t)b * N_ * N_;
    const uint8_t* arow = base + (size_t)(tm + l) * N_;
    const uint8_t* brow = base + (size_t)(tn + l) * N_;

    v8i acc = {};
    for (int k = 0; k < N_; k += 64) {
        const int ao = k + 16 * half;
        int2 c0 = *reinterpret_cast<const int2*>(arow + ao);
        int2 c1 = *reinterpret_cast<const int2*>(arow + ao + 8);
        int2 c2 = *reinterpret_cast<const int2*>(arow + ao + 32);
        int2 c3 = *reinterpret_cast<const int2*>(arow + ao + 40);
        v8i a;
        a[0]=c0.x; a[1]=c0.y; a[2]=c1.x; a[3]=c1.y;
        a[4]=c2.x; a[5]=c2.y; a[6]=c3.x; a[7]=c3.y;

        const int bo = k + 16 * half;
        int4 d0 = *reinterpret_cast<const int4*>(brow + bo);
        int4 d1 = *reinterpret_cast<const int4*>(brow + bo + 32);
        v8i bv;
        bv[0]=d0.x; bv[1]=d0.y; bv[2]=d0.z; bv[3]=d0.w;
        bv[4]=d1.x; bv[5]=d1.y; bv[6]=d1.z; bv[7]=d1.w;

        acc = __builtin_amdgcn_wmma_i32_16x16x64_iu8(false, a, false, bv, acc,
                                                     false, false);
    }
    float* out = adjS + (size_t)b * N_ * N_ + (size_t)tm * N_ + tn + l;
#pragma unroll
    for (int v = 0; v < 8; ++v)
        out[(size_t)(v + 8 * half) * N_] = (float)acc[v];
}

__global__ void k_xpool_wmma_f32(const float* __restrict__ adj,
                                 const float* __restrict__ x,
                                 float* __restrict__ xp) {
    const int lane = threadIdx.x & 31;
    const int wid  = threadIdx.x >> 5;
    const int tm   = blockIdx.x << 4;
    const int tf   = wid << 4;
    const int b    = blockIdx.y;
    const int l    = lane & 15;
    const int half = lane >> 4;

    const float* arow = adj + ((size_t)b * N_ + tm + l) * N_;
    const float* xcol = x + (size_t)b * N_ * F_ + tf + l;

    v8f acc = {};
    for (int k = 0; k < N_; k += 4) {
        float2 av = *reinterpret_cast<const float2*>(arow + k + 2 * half);
        v2f a; a[0] = av.x; a[1] = av.y;
        v2f bvec;
        bvec[0] = xcol[(size_t)(k + 2 * half)     * F_];
        bvec[1] = xcol[(size_t)(k + 2 * half + 1) * F_];
        acc = __builtin_amdgcn_wmma_f32_16x16x4_f32(false, a, false, bvec,
                                                    (short)0, acc, false, false);
    }
    float* out = xp + (size_t)b * N_ * F_ + (size_t)tm * F_ + tf + l;
#pragma unroll
    for (int v = 0; v < 8; ++v)
        out[(size_t)(v + 8 * half) * F_] = acc[v];
}

__global__ void k_greedy(const float* __restrict__ adj, const float* __restrict__ adjS,
                         const int* __restrict__ mask, const float* __restrict__ order,
                         int* __restrict__ selected) {
    const int t = threadIdx.x;
    const int b = blockIdx.x;
    __shared__ float rv[N_];
    __shared__ int   ri[N_];
    __shared__ unsigned char avail[N_], front[N_], sel[N_];
    __shared__ int s_idx;

    const float ord = order[(size_t)b * N_ + t];
    avail[t] = (mask[(size_t)b * N_ + t] != 0);
    sel[t]   = 0;

    rv[t] = ord; ri[t] = t;
    __syncthreads();
    for (int s = 512; s > 0; s >>= 1) {
        if (t < s && (rv[t + s] < rv[t] || (rv[t + s] == rv[t] && ri[t + s] < ri[t]))) {
            rv[t] = rv[t + s]; ri[t] = ri[t + s];
        }
        __syncthreads();
    }
    if (t == 0) s_idx = ri[0];
    __syncthreads();
    front[t] = (t == s_idx);
    __syncthreads();

    for (int iter = 0; iter < 2 * N_; ++iter) {
        ri[t] = avail[t];
        __syncthreads();
        for (int s = 512; s > 0; s >>= 1) { if (t < s) ri[t] |= ri[t + s]; __syncthreads(); }
        const int anyA = ri[0];
        __syncthreads();
        if (!anyA) break;

        ri[t] = front[t];
        __syncthreads();
        for (int s = 512; s > 0; s >>= 1) { if (t < s) ri[t] |= ri[t + s]; __syncthreads(); }
        const int anyF = ri[0];
        __syncthreads();

        const int cidx = s_idx;
        if (t == cidx) sel[t] = 1;
        const float ex  = adj [((size_t)b * N_ + cidx) * N_ + t];
        const float inc = adjS[((size_t)b * N_ + cidx) * N_ + t];
        const unsigned char av = avail[t] && (t != cidx) && (ex == 0.0f);
        avail[t] = av;
        front[t] = ((front[t] || (inc != 0.0f) || !anyF) && av) ? 1 : 0;
        __syncthreads();

        rv[t] = front[t] ? ord : __builtin_inff();
        ri[t] = t;
        __syncthreads();
        for (int s = 512; s > 0; s >>= 1) {
            if (t < s && (rv[t + s] < rv[t] || (rv[t + s] == rv[t] && ri[t + s] < ri[t]))) {
                rv[t] = rv[t + s]; ri[t] = ri[t + s];
            }
            __syncthreads();
        }
        if (t == 0 && rv[0] < __builtin_inff()) s_idx = ri[0];
        __syncthreads();
    }
    selected[(size_t)b * N_ + t] = sel[t];
}

__global__ void k_perm(const int* __restrict__ selected, int* __restrict__ perm,
                       int* __restrict__ Scount, float* __restrict__ pool_mask) {
    const int t = threadIdx.x;
    const int b = blockIdx.x;
    __shared__ int buf[N_];
    const int s = (selected[(size_t)b * N_ + t] != 0);
    buf[t] = s;
    __syncthreads();
    for (int off = 1; off < N_; off <<= 1) {
        int v = (t >= off) ? buf[t - off] : 0;
        __syncthreads();
        buf[t] += v;
        __syncthreads();
    }
    const int S       = buf[N_ - 1];
    const int rankSel = buf[t] - s;
    const int pos     = s ? rankSel : S + (t - rankSel);
    perm[(size_t)b * N_ + pos]    = t;
    pool_mask[(size_t)b * N_ + t] = (t < S) ? 1.0f : 0.0f;
    if (t == 0) Scount[b] = S;
}

__global__ void k_gather_x(const float* __restrict__ xp, const int* __restrict__ perm,
                           const int* __restrict__ S, float* __restrict__ xout) {
    size_t i = (size_t)blockIdx.x * blockDim.x + threadIdx.x;
    if (i >= (size_t)B_ * N_ * F_) return;
    const int f = (int)(i % F_);
    const size_t jn = i / F_;
    const int j = (int)(jn % N_);
    const int b = (int)(jn / N_);
    const int p = perm[(size_t)b * N_ + j];
    const float m = (j < S[b]) ? 1.0f : 0.0f;
    xout[i] = m * xp[((size_t)b * N_ + p) * F_ + f];
}

__global__ void k_gather_adj(const float* __restrict__ adjS, const int* __restrict__ perm,
                             const int* __restrict__ S, float* __restrict__ aout) {
    size_t i = (size_t)blockIdx.x * blockDim.x + threadIdx.x;
    if (i >= (size_t)B_ * N_ * N_) return;
    const int kc = (int)(i % N_);
    const size_t jn = i / N_;
    const int j = (int)(jn % N_);
    const int b = (int)(jn / N_);
    const int Sb = S[b];
    const float m = (j < Sb && kc < Sb) ? 1.0f : 0.0f;
    const int pj = perm[(size_t)b * N_ + j];
    const int pk = perm[(size_t)b * N_ + kc];
    aout[i] = m * adjS[((size_t)b * N_ + pj) * N_ + pk];
}

extern "C" void kernel_launch(void* const* d_in, const int* in_sizes, int n_in,
                              void* d_out, int out_size, void* d_ws, size_t ws_size,
                              hipStream_t stream) {
    const float* x     = (const float*)d_in[0];
    const float* adj   = (const float*)d_in[1];
    const int*   mask  = (const int*)  d_in[2];
    const float* order = (const float*)d_in[3];

    float* xout  = (float*)d_out;
    float* aout  = xout + (size_t)B_ * N_ * F_;
    float* pmask = aout + (size_t)B_ * N_ * N_;

    uint8_t* ws     = (uint8_t*)d_ws;
    uint8_t* adjU8  = ws;
    float*   adjS   = (float*)(ws + (size_t)B_ * N_ * N_);
    float*   xp     = (float*)((uint8_t*)adjS + (size_t)B_ * N_ * N_ * 4);
    int* selected   = (int*)((uint8_t*)xp + (size_t)B_ * N_ * F_ * 4);
    int* perm       = selected + (size_t)B_ * N_;
    int* Sc         = perm + (size_t)B_ * N_;

    {
        const int total4 = B_ * N_ * N_ / 4;
        k_adj_to_u8<<<(total4 + 255) / 256, 256, 0, stream>>>(adj, adjU8, total4);
    }
    k_adj2_wmma_iu8<<<dim3(512, B_), 256, 0, stream>>>(adjU8, adjS);
    k_xpool_wmma_f32<<<dim3(N_ / 16, B_), 256, 0, stream>>>(adj, x, xp);
    k_greedy<<<B_, N_, 0, stream>>>(adj, adjS, mask, order, selected);
    k_perm<<<B_, N_, 0, stream>>>(selected, perm, Sc, pmask);
    {
        const size_t nx = (size_t)B_ * N_ * F_;
        k_gather_x<<<(unsigned)((nx + 255) / 256), 256, 0, stream>>>(xp, perm, Sc, xout);
        const size_t na = (size_t)B_ * N_ * N_;
        k_gather_adj<<<(unsigned)((na + 255) / 256), 256, 0, stream>>>(adjS, perm, Sc, aout);
    }
}
